// AutoRegressive_12128987644588
// MI455X (gfx1250) — matrix operand order repaired and checked by reading, not compiled, not run on hardware
//
#include <hip/hip_runtime.h>

// ---------------------------------------------------------------------------
// Model constants (from reference)
// ---------------------------------------------------------------------------
#define BATCH   4
#define TT_     128
#define TA_     512
#define ENR_    225
#define TTOT    865          // real sequence length
#define TP_     896          // padded: 7*128, multiple of 32
#define DMODEL  1024
#define NHEAD_  16
#define HDIM    64
#define FFDIM   4096
#define LAYERS  6
#define PREFIX_ 640          // TT_ + TA_
#define BT      (BATCH * TTOT)
#define BTP     (BATCH * TP_)

typedef __attribute__((ext_vector_type(16))) _Float16 v16h;
typedef __attribute__((ext_vector_type(8)))  _Float16 v8h;
typedef __attribute__((ext_vector_type(8)))  float    v8f;

// ---------------------------------------------------------------------------
// f16-native WMMA fragment loaders — no bounds checks (padded dims), pure
// vector loads, zero conversion work in the hot loop.
// ---------------------------------------------------------------------------
// A 16x32 f16 fragment: lane row M = m0+(lane&15); two contiguous 8-half runs
// at K offsets hb and hb+8 (hb = (lane>=16)*16) -> 2x 16B loads.
__device__ __forceinline__ v16h load_a_frag(const _Float16* __restrict__ Ap,
                                            int lda, int m0, int k0, int lane) {
    const _Float16* p = Ap + (long)(m0 + (lane & 15)) * lda + k0 + ((lane >> 4) << 4);
    v8h lo = *(const v8h*)(p);
    v8h hi = *(const v8h*)(p + 8);
    v16h r;
#pragma unroll
    for (int e = 0; e < 8; ++e) { r[e] = lo[e]; r[e + 8] = hi[e]; }
    return r;
}

// B 32x16 f16 fragment from W[N,K] row-major: lane column N = n0+(lane&15);
// 16 contiguous halves at K = k0 + (lane>=16)*16 -> one 32B vector load.
__device__ __forceinline__ v16h load_b_frag(const _Float16* __restrict__ Bp,
                                            int ldb, int n0, int k0, int lane) {
    const _Float16* p = Bp + (long)(n0 + (lane & 15)) * ldb + k0 + ((lane >> 4) << 4);
    return *(const v16h*)(p);
}

// ---------------------------------------------------------------------------
// Batched GEMM: C = alpha * A @ W^T + bias (+mask/relu in epilogue).
// Block = 256 thr = 8 waves; wave tile 32x64 -> block tile 128x128.
// Per K-step: 2 A frags + 4 B frags -> 8 v_wmma_f32_16x16x32_f16.
// Epilogue can store f32 (Cf) and/or f16 (Ch).
// maskMode bit0: causal-after-prefix (score -inf if n>m && n>=prefix)
// maskMode bit1: kill padded keys     (score -inf if n>=kValid)
// ---------------------------------------------------------------------------
__global__ __launch_bounds__(256) void gemm_wmma_kernel(
    const _Float16* __restrict__ A, const _Float16* __restrict__ B,
    const float* __restrict__ bias, float* __restrict__ Cf,
    _Float16* __restrict__ Ch,
    int M, int N, int K, int lda, int ldb, int ldc,
    long aPB, long aPH, long bPB, long bPH, long cPB, long cPH,
    int H, float alpha, int maskMode, int prefix, int kValid, int doRelu) {

    int z  = blockIdx.z;
    int pb = z / H;
    int ph = z - pb * H;
    const _Float16* Ap = A + (long)pb * aPB + (long)ph * aPH;
    const _Float16* Bp = B + (long)pb * bPB + (long)ph * bPH;

    int lane = threadIdx.x & 31;
    int w    = threadIdx.x >> 5;
    int m0   = blockIdx.y * 128 + (w & 3) * 32;   // wave M origin
    int n0   = blockIdx.x * 128 + (w >> 2) * 64;  // wave N origin

    v8f acc[2][4];
#pragma unroll
    for (int i = 0; i < 2; ++i)
#pragma unroll
        for (int j = 0; j < 4; ++j)
            acc[i][j] = (v8f){0.f, 0.f, 0.f, 0.f, 0.f, 0.f, 0.f, 0.f};

    for (int kk = 0; kk < K; kk += 32) {
        v16h a0 = load_a_frag(Ap, lda, m0,      kk, lane);
        v16h a1 = load_a_frag(Ap, lda, m0 + 16, kk, lane);
#pragma unroll
        for (int j = 0; j < 4; ++j) {
            v16h b = load_b_frag(Bp, ldb, n0 + 16 * j, kk, lane);
            acc[0][j] = __builtin_amdgcn_wmma_f32_16x16x32_f16(
                false, a0, false, b, (short)0, acc[0][j], false, false);
            acc[1][j] = __builtin_amdgcn_wmma_f32_16x16x32_f16(
                false, a1, false, b, (short)0, acc[1][j], false, false);
        }
    }

    // C/D layout: n = lane&15; m = m0 + 16*i + 8*(lane>>4) + vgpr
    long cOff = (long)pb * cPB + (long)ph * cPH;
    int cn = lane & 15;
    int rs = (lane >> 4) * 8;
#pragma unroll
    for (int i = 0; i < 2; ++i) {
#pragma unroll
        for (int j = 0; j < 4; ++j) {
            int nn = n0 + 16 * j + cn;
            if (nn >= N) continue;
            float bv = bias ? bias[nn] : 0.f;
#pragma unroll
            for (int v = 0; v < 8; ++v) {
                int mm = m0 + 16 * i + rs + v;
                if (mm >= M) continue;
                float val = acc[i][j][v] * alpha + bv;
                if (maskMode & 1) { if (nn > mm && nn >= prefix) val = -1e30f; }
                if (maskMode & 2) { if (nn >= kValid) val = -1e30f; }
                if (doRelu) val = fmaxf(val, 0.f);
                long idx = cOff + (long)mm * ldc + nn;
                if (Cf) Cf[idx] = val;
                if (Ch) Ch[idx] = (_Float16)val;  // -1e30 -> -inf in f16
            }
        }
    }
}

// ---------------------------------------------------------------------------
// Elementwise f32 -> f16 (weights, once per launch)
// ---------------------------------------------------------------------------
__global__ __launch_bounds__(256) void cvt_f16_kernel(
    const float* __restrict__ src, _Float16* __restrict__ dst, long n) {
    long i = (long)blockIdx.x * 256 + threadIdx.x;
    if (i < n) dst[i] = (_Float16)src[i];
}

// ---------------------------------------------------------------------------
// Transpose V out of the fused f16 QKV buffer:
//   VT[((b*H + h)*HDIM + e)*TP_ + k] = QKV[(b*TP_ + k)*3072 + 2048 + h*64 + e]
// ---------------------------------------------------------------------------
__global__ __launch_bounds__(256) void transpose_v_kernel(
    const _Float16* __restrict__ QKV, _Float16* __restrict__ VT) {
    long idx = (long)blockIdx.x * 256 + threadIdx.x;   // B*TP_*DMODEL total
    int  c = (int)(idx & (DMODEL - 1));
    long t = idx >> 10;
    int  k = (int)(t % TP_);
    int  b = (int)(t / TP_);
    int  h = c >> 6;
    int  e = c & 63;
    VT[((long)(b * NHEAD_ + h) * HDIM + e) * TP_ + k] =
        QKV[((long)b * TP_ + k) * (3 * DMODEL) + 2 * DMODEL + c];
}

// ---------------------------------------------------------------------------
// Row softmax over f16 scores, in place (f32 math). Masked entries are -inf
// in f16 and become exactly 0.
// ---------------------------------------------------------------------------
__global__ __launch_bounds__(256) void softmax_rows_kernel(
    _Float16* __restrict__ S, int T) {
    long row = blockIdx.x;
    _Float16* p = S + row * (long)T;
    __shared__ float red[256];
    int tid = threadIdx.x;

    float mx = -3.4e38f;
    for (int i = tid; i < T; i += 256) mx = fmaxf(mx, (float)p[i]);
    red[tid] = mx; __syncthreads();
    for (int s = 128; s > 0; s >>= 1) {
        if (tid < s) red[tid] = fmaxf(red[tid], red[tid + s]);
        __syncthreads();
    }
    mx = red[0]; __syncthreads();

    float sum = 0.f;
    for (int i = tid; i < T; i += 256) {
        float e = __expf((float)p[i] - mx);
        p[i] = (_Float16)e;
        sum += e;
    }
    red[tid] = sum; __syncthreads();
    for (int s = 128; s > 0; s >>= 1) {
        if (tid < s) red[tid] += red[tid + s];
        __syncthreads();
    }
    float inv = 1.f / red[0];
    for (int i = tid; i < T; i += 256) p[i] = (_Float16)((float)p[i] * inv);
}

// ---------------------------------------------------------------------------
// X = LayerNorm(x + h) * w + b  -> writes f32 X and f16 X16.
// Safe for out == x (each element touched by exactly one thread).
// ---------------------------------------------------------------------------
__global__ __launch_bounds__(256) void add_ln_kernel(
    const float* __restrict__ x, const float* __restrict__ h,
    const float* __restrict__ w, const float* __restrict__ b,
    float* __restrict__ out, _Float16* __restrict__ out16) {
    long base = (long)blockIdx.x * DMODEL;
    int tid = threadIdx.x;
    __shared__ float red[256];

    float v[4];
    float s = 0.f;
#pragma unroll
    for (int j = 0; j < 4; ++j) {
        int c = tid + j * 256;
        v[j] = x[base + c] + h[base + c];
        s += v[j];
    }
    red[tid] = s; __syncthreads();
    for (int st = 128; st > 0; st >>= 1) {
        if (tid < st) red[tid] += red[tid + st];
        __syncthreads();
    }
    float mean = red[0] * (1.f / DMODEL);
    __syncthreads();

    float vs = 0.f;
#pragma unroll
    for (int j = 0; j < 4; ++j) {
        float d = v[j] - mean;
        vs += d * d;
    }
    red[tid] = vs; __syncthreads();
    for (int st = 128; st > 0; st >>= 1) {
        if (tid < st) red[tid] += red[tid + st];
        __syncthreads();
    }
    float inv = rsqrtf(red[0] * (1.f / DMODEL) + 1e-5f);
    __syncthreads();

#pragma unroll
    for (int j = 0; j < 4; ++j) {
        int c = tid + j * 256;
        float o = (v[j] - mean) * inv * w[c] + b[c];
        out[base + c]   = o;
        out16[base + c] = (_Float16)o;
    }
}

// ---------------------------------------------------------------------------
// Embedding + sinusoidal PE + pack; pad rows (pos >= TTOT) become zeros.
// Writes f32 X (residual stream) and f16 copies X16 / MEM16 (GEMM inputs).
// ---------------------------------------------------------------------------
__global__ __launch_bounds__(256) void embed_pack_kernel(
    const int* __restrict__ text, const int* __restrict__ audio,
    const int* __restrict__ enrolled, const int* __restrict__ tlen,
    const int* __restrict__ alen, const float* __restrict__ temb,
    const float* __restrict__ aemb, float* __restrict__ X,
    _Float16* __restrict__ X16, _Float16* __restrict__ MEM16) {
    int row = blockIdx.x;
    int b   = row / TP_;
    int pos = row - b * TP_;
    int tl = tlen[b];
    int al = alen[b];

    const float* src = nullptr;
    int p = 0;
    if (pos < tl) {
        p = pos;
        src = temb + (long)text[b * TT_ + p] * DMODEL;
    } else if (pos < tl + al) {
        p = pos - tl;
        src = aemb + (long)audio[b * TA_ + p] * DMODEL;
    } else if (pos < tl + al + ENR_) {
        p = pos - tl - al;
        src = aemb + (long)enrolled[b * ENR_ + p] * DMODEL;
    }

    long base = (long)row * DMODEL;
    const float kLog = 9.2103403719761836f / (float)DMODEL;  // ln(10000)/D
    for (int c = threadIdx.x; c < DMODEL; c += 256) {
        float val = 0.f;
        if (src) {
            float f   = __expf(-(float)(c & ~1) * kLog);
            float ang = (float)p * f;
            float pe  = (c & 1) ? __cosf(ang) : __sinf(ang);
            val = src[c] + pe;
        }
        X[base + c]     = val;
        X16[base + c]   = (_Float16)val;
        MEM16[base + c] = (_Float16)val;
    }
}

// ---------------------------------------------------------------------------
// Host-side launcher
// ---------------------------------------------------------------------------
static void launch_gemm(hipStream_t st, const _Float16* A, const _Float16* B,
                        const float* bias, float* Cf, _Float16* Ch, int M,
                        int Mgrid, int N, int K, int lda, int ldb, int ldc,
                        long aPB, long aPH, long bPB, long bPH, long cPB,
                        long cPH, int planes, int H, float alpha, int maskMode,
                        int prefix, int kValid, int relu) {
    dim3 grid((N + 127) / 128, (Mgrid + 127) / 128, planes);
    gemm_wmma_kernel<<<grid, 256, 0, st>>>(A, B, bias, Cf, Ch, M, N, K, lda,
                                           ldb, ldc, aPB, aPH, bPB, bPH, cPB,
                                           cPH, H, alpha, maskMode, prefix,
                                           kValid, relu);
}

static void launch_cvt(hipStream_t st, const float* src, _Float16* dst, long n) {
    cvt_f16_kernel<<<(unsigned)((n + 255) / 256), 256, 0, st>>>(src, dst, n);
}

extern "C" void kernel_launch(void* const* d_in, const int* in_sizes, int n_in,
                              void* d_out, int out_size, void* d_ws,
                              size_t ws_size, hipStream_t stream) {
    const int*   text     = (const int*)d_in[0];
    const int*   audio    = (const int*)d_in[1];
    const int*   enrolled = (const int*)d_in[2];
    const int*   tlen     = (const int*)d_in[3];
    const int*   alen     = (const int*)d_in[4];
    const float* temb     = (const float*)d_in[5];
    const float* aemb     = (const float*)d_in[6];
    const float* sa_in_w  = (const float*)d_in[7];
    const float* sa_in_b  = (const float*)d_in[8];
    const float* sa_out_w = (const float*)d_in[9];
    const float* sa_out_b = (const float*)d_in[10];
    const float* ca_in_w  = (const float*)d_in[11];
    const float* ca_in_b  = (const float*)d_in[12];
    const float* ca_out_w = (const float*)d_in[13];
    const float* ca_out_b = (const float*)d_in[14];
    const float* ff1_w    = (const float*)d_in[15];
    const float* ff1_b    = (const float*)d_in[16];
    const float* ff2_w    = (const float*)d_in[17];
    const float* ff2_b    = (const float*)d_in[18];
    const float* ln1_w    = (const float*)d_in[19];
    const float* ln1_b    = (const float*)d_in[20];
    const float* ln2_w    = (const float*)d_in[21];
    const float* ln2_b    = (const float*)d_in[22];
    const float* ln3_w    = (const float*)d_in[23];
    const float* ln3_b    = (const float*)d_in[24];
    const float* out_w    = (const float*)d_in[25];
    const float* out_b    = (const float*)d_in[26];
    (void)in_sizes; (void)n_in; (void)out_size; (void)ws_size;

    // Workspace carve-out (activations padded to TP_ rows/batch)
    char* ws = (char*)d_ws;
    auto carve = [&](size_t bytes) -> char* {
        char* p = ws;
        ws += (bytes + 255) & ~(size_t)255;
        return p;
    };
    const int D = DMODEL, H = NHEAD_;

    float*     X     = (float*)carve((size_t)BTP * D * 4);
    float*     T1    = (float*)carve((size_t)BTP * D * 4);
    _Float16*  X16   = (_Float16*)carve((size_t)BTP * D * 2);
    _Float16*  MEM16 = (_Float16*)carve((size_t)BTP * D * 2);
    _Float16*  QKV16 = (_Float16*)carve((size_t)BTP * 3 * D * 2);
    _Float16*  HFF16 = (_Float16*)carve((size_t)BTP * FFDIM * 2);
    _Float16*  OAT16 = (_Float16*)carve((size_t)BTP * D * 2);
    // VT: [B*H][HDIM][TP_] + 64 pad rows for the unused N=[64,128) half-tile
    _Float16*  VT16  = (_Float16*)carve(((size_t)BATCH * H * HDIM + 64) * TP_ * 2);
    _Float16*  S16   = (_Float16*)carve((size_t)BATCH * H * TP_ * TP_ * 2);
    // f16 weights (converted once)
    _Float16*  saW16  = (_Float16*)carve((size_t)LAYERS * 3 * D * D * 2);
    _Float16*  saOW16 = (_Float16*)carve((size_t)LAYERS * D * D * 2);
    _Float16*  caW16  = (_Float16*)carve((size_t)LAYERS * 3 * D * D * 2);
    _Float16*  caOW16 = (_Float16*)carve((size_t)LAYERS * D * D * 2);
    _Float16*  f1W16  = (_Float16*)carve((size_t)LAYERS * FFDIM * D * 2);
    _Float16*  f2W16  = (_Float16*)carve((size_t)LAYERS * D * FFDIM * 2);
    _Float16*  oW16   = (_Float16*)carve((size_t)D * D * 2);

    const long TT3 = (long)TP_ * 3 * D;     // per-batch stride in QKV
    const long STT = (long)TP_ * TP_;       // per-head stride in S
    const long VTB = (long)H * HDIM * TP_;  // per-batch stride in VT
    const long VTH = (long)HDIM * TP_;      // per-head stride in VT
    const int  TRB = (BATCH * TP_ * DMODEL) / 256;  // transpose grid

    // 0) weights -> f16 (once per launch)
    launch_cvt(stream, sa_in_w,  saW16,  (long)LAYERS * 3 * D * D);
    launch_cvt(stream, sa_out_w, saOW16, (long)LAYERS * D * D);
    launch_cvt(stream, ca_in_w,  caW16,  (long)LAYERS * 3 * D * D);
    launch_cvt(stream, ca_out_w, caOW16, (long)LAYERS * D * D);
    launch_cvt(stream, ff1_w,    f1W16,  (long)LAYERS * FFDIM * D);
    launch_cvt(stream, ff2_w,    f2W16,  (long)LAYERS * D * FFDIM);
    launch_cvt(stream, out_w,    oW16,   (long)D * D);

    // 1) embed + pack -> X (f32), X16, MEM16
    embed_pack_kernel<<<BATCH * TP_, 256, 0, stream>>>(text, audio, enrolled,
                                                       tlen, alen, temb, aemb,
                                                       X, X16, MEM16);

    for (int l = 0; l < LAYERS; ++l) {
        // ---- self attention ----
        launch_gemm(stream, X16, saW16 + (long)l * 3 * D * D,
                    sa_in_b + (long)l * 3 * D, nullptr, QKV16, BTP, BTP,
                    3 * D, D, D, D, 3 * D, 0, 0, 0, 0, 0, 0, 1, 1,
                    1.f, 0, 0, 0, 0);
        transpose_v_kernel<<<TRB, 256, 0, stream>>>(QKV16, VT16);
        // scores = Q K^T / 8 (causal-after-prefix + kill padded keys) -> f16
        launch_gemm(stream, QKV16, QKV16 + D, nullptr, nullptr, S16, TP_, TP_,
                    TP_, HDIM, 3 * D, 3 * D, TP_, TT3, HDIM, TT3, HDIM,
                    (long)H * STT, STT, BATCH * H, H, 0.125f, 3, PREFIX_,
                    TTOT, 0);
        softmax_rows_kernel<<<BATCH * H * TP_, 256, 0, stream>>>(S16, TP_);
        // context = P V
        launch_gemm(stream, S16, VT16, nullptr, nullptr, OAT16, TP_, TP_,
                    HDIM, TP_, TP_, TP_, D, (long)H * STT, STT, VTB, VTH,
                    (long)TP_ * D, HDIM, BATCH * H, H, 1.f, 0, 0, 0, 0);
        // output projection -> f32, then residual + LN1
        launch_gemm(stream, OAT16, saOW16 + (long)l * D * D,
                    sa_out_b + (long)l * D, T1, nullptr, BTP, BTP, D, D,
                    D, D, D, 0, 0, 0, 0, 0, 0, 1, 1, 1.f, 0, 0, 0, 0);
        add_ln_kernel<<<BTP, 256, 0, stream>>>(X, T1, ln1_w + (long)l * D,
                                               ln1_b + (long)l * D, X, X16);

        // ---- cross attention (Q from X, K/V from MEM) ----
        launch_gemm(stream, X16, caW16 + (long)l * 3 * D * D,
                    ca_in_b + (long)l * 3 * D, nullptr, QKV16, BTP, BTP,
                    D, D, D, D, 3 * D, 0, 0, 0, 0, 0, 0, 1, 1,
                    1.f, 0, 0, 0, 0);
        launch_gemm(stream, MEM16, caW16 + (long)l * 3 * D * D + (long)D * D,
                    ca_in_b + (long)l * 3 * D + D, nullptr, QKV16 + D, BTP,
                    BTP, 2 * D, D, D, D, 3 * D, 0, 0, 0, 0, 0, 0, 1, 1,
                    1.f, 0, 0, 0, 0);
        transpose_v_kernel<<<TRB, 256, 0, stream>>>(QKV16, VT16);
        launch_gemm(stream, QKV16, QKV16 + D, nullptr, nullptr, S16, TP_, TP_,
                    TP_, HDIM, 3 * D, 3 * D, TP_, TT3, HDIM, TT3, HDIM,
                    (long)H * STT, STT, BATCH * H, H, 0.125f, 2, 0, TTOT, 0);
        softmax_rows_kernel<<<BATCH * H * TP_, 256, 0, stream>>>(S16, TP_);
        launch_gemm(stream, S16, VT16, nullptr, nullptr, OAT16, TP_, TP_,
                    HDIM, TP_, TP_, TP_, D, (long)H * STT, STT, VTB, VTH,
                    (long)TP_ * D, HDIM, BATCH * H, H, 1.f, 0, 0, 0, 0);
        launch_gemm(stream, OAT16, caOW16 + (long)l * D * D,
                    ca_out_b + (long)l * D, T1, nullptr, BTP, BTP, D, D,
                    D, D, D, 0, 0, 0, 0, 0, 0, 1, 1, 1.f, 0, 0, 0, 0);
        add_ln_kernel<<<BTP, 256, 0, stream>>>(X, T1, ln2_w + (long)l * D,
                                               ln2_b + (long)l * D, X, X16);

        // ---- feed forward ----
        launch_gemm(stream, X16, f1W16 + (long)l * FFDIM * D,
                    ff1_b + (long)l * FFDIM, nullptr, HFF16, BTP, BTP,
                    FFDIM, D, D, D, FFDIM, 0, 0, 0, 0, 0, 0, 1, 1,
                    1.f, 0, 0, 0, /*relu*/ 1);
        launch_gemm(stream, HFF16, f2W16 + (long)l * D * FFDIM,
                    ff2_b + (long)l * D, T1, nullptr, BTP, BTP, D, FFDIM,
                    FFDIM, FFDIM, D, 0, 0, 0, 0, 0, 0, 1, 1, 1.f, 0, 0, 0, 0);
        add_ln_kernel<<<BTP, 256, 0, stream>>>(X, T1, ln3_w + (long)l * D,
                                               ln3_b + (long)l * D, X, X16);
    }

    // final logits: d_out = X @ out_w^T + out_b  (only real rows stored)
    launch_gemm(stream, X16, oW16, out_b, (float*)d_out, nullptr, BT, BTP,
                1024, D, D, D, 1024, 0, 0, 0, 0, 0, 0, 1, 1, 1.f, 0, 0, 0, 0);
}
